// GAT_Layer_68350109549109
// MI455X (gfx1250) — matrix operand order repaired and checked by reading, not compiled, not run on hardware
//
#include <hip/hip_runtime.h>

#define BB 4
#define N1 2048
#define NROW 2047
#define II 256
#define OO 128
#define HH 4
#define NEGV (-1e18f)
#define SLOPEV 0.2f

typedef __attribute__((ext_vector_type(16))) __bf16 v16bf;
typedef __attribute__((ext_vector_type(8)))  __bf16 v8bf;
typedef __attribute__((ext_vector_type(8)))  float  v8f;
typedef __attribute__((ext_vector_type(4)))  float  v4f;
typedef __attribute__((ext_vector_type(4)))  unsigned int u32x4;
typedef __attribute__((ext_vector_type(8)))  int          i32x8;
typedef __attribute__((ext_vector_type(4)))  int          i32x4;

static __device__ __forceinline__ v16bf cat8(v8bf lo, v8bf hi) {
  return __builtin_shufflevector(lo, hi, 0,1,2,3,4,5,6,7,8,9,10,11,12,13,14,15);
}

// ---------------------------------------------------------------------------
// TDM: 1-D tile load (8192 bytes, data_size = 2B, tile_dim0 = 4096 elems)
// D# group0: count=1 | lds_addr | global_addr[56:0] | type=2
// D# group1: data_size=1(2B); tensor_dim0=1<<30; tensor_dim1=1; tile_dim0=4096
// (6-arg toolchain variant: extra int32x8 group before cache-policy imm)
// ---------------------------------------------------------------------------
static __device__ __forceinline__ void tdm_load_tile(unsigned int lds_byte_addr,
                                                     const __bf16* gsrc) {
  const unsigned long long ga = (unsigned long long)(size_t)gsrc;
  u32x4 g0 = { 1u,                                   // count=1 (user mode)
               lds_byte_addr,                        // lds_addr
               (unsigned int)ga,                     // global_addr[31:0]
               (unsigned int)((ga >> 32) & 0x01FFFFFFull) | (2u << 30) }; // +type=2
  i32x8 g1 = { 0x00010000,          // workgroup_mask=0, data_size=1 (2 bytes)
               0,                    // tensor_dim0[15:0] in [31:16] (1<<30 -> 0)
               0x00014000,           // tensor_dim0[31:16]=0x4000 ; tensor_dim1 lo16=1
               (int)(4096u << 16),   // tile_dim0=4096 in [31:16]; tensor_dim1 hi16=0
               0,                    // tile_dim1=0 (1-D tile), tile_dim2=0
               4096,                 // tensor_dim0_stride low32
               0, 0 };
  i32x4 g2 = { 0, 0, 0, 0 };
  i32x4 g3 = { 0, 0, 0, 0 };
  i32x8 g4 = { 0, 0, 0, 0, 0, 0, 0, 0 };
  __builtin_amdgcn_tensor_load_to_lds(g0, g1, g2, g3, g4, 0);
}

// ---------------------------------------------------------------------------
// K0: zero s_src/s_dst accumulators (contiguous)
// ---------------------------------------------------------------------------
__global__ void k_zero(float* __restrict__ p, int n) {
  int i = blockIdx.x * blockDim.x + threadIdx.x;
  if (i < n) p[i] = 0.0f;
}

// ---------------------------------------------------------------------------
// K0b: initialize output with broadcast bias; k_attn atomically accumulates
// ---------------------------------------------------------------------------
__global__ void k_initout(float* __restrict__ out, const float* __restrict__ bias, int n) {
  int i = blockIdx.x * blockDim.x + threadIdx.x;
  if (i < n) out[i] = bias[i & (OO - 1)];
}

// ---------------------------------------------------------------------------
// K0c: pre-transpose w_head into blocked bf16 WMMA-B layout:
// whT[(((h*4+kt)*8+cb)*2+half)*256 + l16*16 + e] = wh[h][kt*32+8*half+(e<8?e:e+8)][cb*16+l16]
// ---------------------------------------------------------------------------
__global__ void k_wht(const float* __restrict__ wh, __bf16* __restrict__ whT) {
  const int idx = blockIdx.x * blockDim.x + threadIdx.x;   // [0, H*OO*OO)
  const int e    = idx & 15;
  const int l16  = (idx >> 4) & 15;
  const int half = (idx >> 8) & 1;
  const int cb   = (idx >> 9) & 7;
  const int kt   = (idx >> 12) & 3;
  const int h    = idx >> 14;
  const int k = kt * 32 + 8 * half + (e < 8 ? e : e + 8);
  const int n = cb * 16 + l16;
  whT[idx] = (__bf16)wh[((size_t)h * OO + k) * OO + n];
}

// ---------------------------------------------------------------------------
// K1: hp[b,n,o] = x[b,n,:] @ W_lin[o,:]   (bf16 WMMA, fp32 accum)
// one wave -> 16 rows x 64 cols; K loop 256/32 = 8; 32 WMMAs / wave
// ---------------------------------------------------------------------------
__global__ void __launch_bounds__(128) k_lin(const float* __restrict__ x,
                                             const float* __restrict__ Wl,
                                             __bf16* __restrict__ hp) {
  const int wave = blockIdx.x * 4 + (threadIdx.x >> 5);
  const int lane = threadIdx.x & 31;
  const int half = lane >> 4, l16 = lane & 15;
  const int cg = wave & 1;
  const int rt = (wave >> 1) & 127;
  const int b  = wave >> 8;
  const int row = rt * 16 + l16;
  const bool rowOK = row < NROW;
  const float* xrow = x + ((size_t)b * NROW + (rowOK ? row : 0)) * II;
  const v4f z4 = {0.f, 0.f, 0.f, 0.f};
  v8f acc[4] = {};
  for (int k0 = 0; k0 < II; k0 += 32) {
    v16bf a;
    {
      const float* p0 = xrow + k0 + 8 * half;
      v4f f0 = rowOK ? *(const v4f*)(p0)      : z4;
      v4f f1 = rowOK ? *(const v4f*)(p0 + 4)  : z4;
      v4f f2 = rowOK ? *(const v4f*)(p0 + 16) : z4;
      v4f f3 = rowOK ? *(const v4f*)(p0 + 20) : z4;
      #pragma unroll
      for (int t = 0; t < 4; ++t) {
        a[t]      = (__bf16)f0[t];
        a[4 + t]  = (__bf16)f1[t];
        a[8 + t]  = (__bf16)f2[t];
        a[12 + t] = (__bf16)f3[t];
      }
    }
    #pragma unroll
    for (int cb = 0; cb < 4; ++cb) {
      const int n = cg * 64 + cb * 16 + l16;
      const float* wr = Wl + (size_t)n * II + k0 + 8 * half;
      v16bf bb;
      #pragma unroll
      for (int q = 0; q < 4; ++q) {
        v4f f = *(const v4f*)(wr + q * 4 + (q >> 1) * 8);
        bb[q * 4 + 0] = (__bf16)f[0];
        bb[q * 4 + 1] = (__bf16)f[1];
        bb[q * 4 + 2] = (__bf16)f[2];
        bb[q * 4 + 3] = (__bf16)f[3];
      }
      acc[cb] = __builtin_amdgcn_wmma_f32_16x16x32_bf16(false, a, false, bb,
                                                        (short)0, acc[cb], false, false);
    }
  }
  #pragma unroll
  for (int cb = 0; cb < 4; ++cb) {
    #pragma unroll
    for (int r = 0; r < 8; ++r) {
      const int orow = rt * 16 + r + 8 * half;
      hp[((size_t)b * N1 + orow) * OO + cg * 64 + cb * 16 + l16] = (__bf16)acc[cb][r];
    }
  }
}

// ---------------------------------------------------------------------------
// K1b: append prior_feature as row 2047
// ---------------------------------------------------------------------------
__global__ void k_prior(const float* __restrict__ prior, __bf16* __restrict__ hp) {
  const int b = blockIdx.x, o = threadIdx.x;
  hp[((size_t)b * N1 + NROW) * OO + o] = (__bf16)prior[b * OO + o];
}

// ---------------------------------------------------------------------------
// K2: hp2t = blocked-layout (hp @ w_head[h]); fused tanh + s_src/s_dst dots
// one wave -> 16x16 tile; 4 WMMAs; single b128 store of the C tile (blocked)
// ---------------------------------------------------------------------------
__global__ void __launch_bounds__(128) k_head(const __bf16* __restrict__ hp,
                                              const __bf16* __restrict__ whT,
                                              const float* __restrict__ asrc,
                                              const float* __restrict__ adst,
                                              __bf16* __restrict__ hp2t,
                                              float* __restrict__ ssrc,
                                              float* __restrict__ sdst) {
  const int wave = blockIdx.x * 4 + (threadIdx.x >> 5);
  const int lane = threadIdx.x & 31;
  const int half = lane >> 4, l16 = lane & 15;
  const int ct = wave & 7;
  const int rt = (wave >> 3) & 127;
  const int h  = (wave >> 10) & 3;
  const int b  = wave >> 12;
  const int row = rt * 16 + l16;
  const int n  = ct * 16 + l16;
  const __bf16* arow = hp + ((size_t)b * N1 + row) * OO;
  v8f c = {};
  #pragma unroll
  for (int kt = 0; kt < 4; ++kt) {
    const int k0 = kt * 32;
    v8bf lo = *(const v8bf*)(arow + k0 + 8 * half);
    v8bf hi = *(const v8bf*)(arow + k0 + 16 + 8 * half);
    v16bf a = cat8(lo, hi);
    const v16bf bb = *(const v16bf*)(whT
        + ((((size_t)h * 4 + kt) * 8 + ct) * 2 + half) * 256 + l16 * 16);
    c = __builtin_amdgcn_wmma_f32_16x16x32_bf16(false, a, false, bb, (short)0, c, false, false);
  }
  // blocked store: tile jt = rt>>1, K-half hh = rt&1, ee = r + 8*half contiguous
  {
    v8bf o;
    #pragma unroll
    for (int r = 0; r < 8; ++r) o[r] = (__bf16)c[r];
    const size_t tbase = ((((size_t)b * HH + h) * 64) + (rt >> 1)) * 4096
                       + ((size_t)(ct * 2 + (rt & 1)) * 16 + l16) * 16 + 8 * half;
    *(v8bf*)(hp2t + tbase) = o;
  }
  const float as = asrc[h * OO + n];
  const float ad = adst[h * OO + n];
  #pragma unroll
  for (int r = 0; r < 8; ++r) {
    const int orow = rt * 16 + r + 8 * half;
    const float t = tanhf(c[r]);
    float vs = t * as, vd = t * ad;
    vs += __shfl_xor(vs, 1); vs += __shfl_xor(vs, 2);
    vs += __shfl_xor(vs, 4); vs += __shfl_xor(vs, 8);
    vd += __shfl_xor(vd, 1); vd += __shfl_xor(vd, 2);
    vd += __shfl_xor(vd, 4); vd += __shfl_xor(vd, 8);
    if (l16 == 0) {
      const size_t ridx = ((size_t)b * HH + h) * N1 + orow;
      atomicAdd(&ssrc[ridx], vs);
      atomicAdd(&sdst[ridx], vd);
    }
  }
}

// ---------------------------------------------------------------------------
// K3a: fold column mask into s_dst
// ---------------------------------------------------------------------------
__global__ void k_sdx(const float* __restrict__ sdst,
                      const unsigned char* __restrict__ xm,
                      float* __restrict__ sdx) {
  const int idx = blockIdx.x * blockDim.x + threadIdx.x;
  const int j = idx & (N1 - 1);
  const int b = idx >> 13;
  sdx[idx] = xm[b * N1 + j] ? -1e30f : sdst[idx];
}

// ---------------------------------------------------------------------------
// K3b: per-row softmax stats (max m, 1/Z). One wave per (b,h,i) row.
// ---------------------------------------------------------------------------
__global__ void __launch_bounds__(128) k_stats(const float* __restrict__ ssrc,
                                               const float* __restrict__ sdx,
                                               const unsigned char* __restrict__ xm,
                                               float* __restrict__ mrow,
                                               float* __restrict__ rzrow) {
  const int wave = blockIdx.x * 4 + (threadIdx.x >> 5);
  const int lane = threadIdx.x & 31;
  const int i  = wave & (N1 - 1);
  const int bh = wave >> 11;
  const int b  = bh >> 2;
  const bool mi = xm[b * N1 + i] != 0;
  if (mi) { // fully-masked row -> softmax over equal NEG -> uniform 1/N1
    if (lane == 0) { mrow[wave] = NEGV; rzrow[wave] = 1.0f / (float)N1; }
    return;
  }
  const float si = ssrc[wave];
  const float* sd = sdx + (size_t)bh * N1;
  float m = -3.0e38f;
  for (int j = lane; j < N1; j += 32) {
    float s = si + sd[j];
    s = s >= 0.f ? s : SLOPEV * s;
    m = fmaxf(m, s);
  }
  m = fmaxf(m, __shfl_xor(m, 1));  m = fmaxf(m, __shfl_xor(m, 2));
  m = fmaxf(m, __shfl_xor(m, 4));  m = fmaxf(m, __shfl_xor(m, 8));
  m = fmaxf(m, __shfl_xor(m, 16));
  float z = 0.f;
  for (int j = lane; j < N1; j += 32) {
    float s = si + sd[j];
    s = s >= 0.f ? s : SLOPEV * s;
    z += __expf(s - m);
  }
  z += __shfl_xor(z, 1); z += __shfl_xor(z, 2);
  z += __shfl_xor(z, 4); z += __shfl_xor(z, 8); z += __shfl_xor(z, 16);
  if (lane == 0) { mrow[wave] = m; rzrow[wave] = 1.0f / z; }
}

// ---------------------------------------------------------------------------
// K4: out += (1/H) * softmax(scores) @ hp2 ; flash-style.
// V tiles (blocked 8 KB) streamed into double-buffered LDS by the Tensor
// Data Mover (wave 0 issues tensor_load_to_lds, syncs via s_wait_tensorcnt).
// B reads are single ds b128x2 (v16bf) per 16x16 tile; weights in-register.
// ---------------------------------------------------------------------------
__global__ void __launch_bounds__(128) k_attn(const __bf16* __restrict__ hp2t,
                                              const float* __restrict__ ssrc,
                                              const float* __restrict__ sdx,
                                              const unsigned char* __restrict__ xm,
                                              const float* __restrict__ mrow,
                                              const float* __restrict__ rzrow,
                                              float* __restrict__ out) {
  __shared__ __attribute__((aligned(32))) __bf16 lb[2][32 * OO]; // 2 x 8 KiB
  const int tid  = threadIdx.x;
  const int lane = tid & 31;
  const int half = lane >> 4, l16 = lane & 15;
  const int wib  = tid >> 5;
  const int h   = blockIdx.x & 3;
  const int rtg = (blockIdx.x >> 2) & 31;
  const int b   = blockIdx.x >> 7;
  const int rt  = rtg * 4 + wib;
  const int rowBase = rt * 16;
  const int i = rowBase + l16;
  const bool mi = xm[b * N1 + i] != 0;

  const size_t rbase = ((size_t)b * HH + h) * N1;
  const float si = ssrc[rbase + i];
  const float m  = mrow[rbase + i];
  const float rz = rzrow[rbase + i];
  const float* sd = sdx + rbase;
  const __bf16* vbase = hp2t + rbase * OO;   // 64 tiles x 4096 bf16 (blocked)

  const bool issuer = (wib == 0);
  const unsigned int lds0 = (unsigned int)(size_t)&lb[0][0];
  const unsigned int lds1 = (unsigned int)(size_t)&lb[1][0];

  if (issuer) tdm_load_tile(lds0, vbase);     // prologue: tile 0 -> buf0

  v8f c[8] = {};
  for (int jt = 0; jt < N1 / 32; ++jt) {
    if (issuer) {
      if (jt < N1 / 32 - 1) {
        tdm_load_tile((jt & 1) ? lds0 : lds1, vbase + (size_t)(jt + 1) * (32 * OO));
        __builtin_amdgcn_s_wait_tensorcnt(1);  // tile jt landed
      } else {
        __builtin_amdgcn_s_wait_tensorcnt(0);
      }
    }
    __syncthreads();
    const __bf16* tile = &lb[jt & 1][0];
    const int j0 = jt * 32;
    __builtin_prefetch(sd + j0 + 32, 0, 1);
    // A tile = softmax weights, contiguous k = e + 16*half (same order as the B tile)
    v16bf a;
    #pragma unroll
    for (int e = 0; e < 16; ++e) {
      const int j = j0 + e + 16 * half;
      float s = si + sd[j];
      s = s >= 0.f ? s : SLOPEV * s;
      const float w = mi ? rz : __expf(s - m) * rz;
      a[e] = (__bf16)w;
    }
    #pragma unroll
    for (int cb = 0; cb < 8; ++cb) {
      const v16bf bb = *(const v16bf*)(tile + ((size_t)(cb * 2 + half) * 16 + l16) * 16);
      c[cb] = __builtin_amdgcn_wmma_f32_16x16x32_bf16(false, a, false, bb,
                                                      (short)0, c[cb], false, false);
    }
    __syncthreads();  // all waves done with buf[jt&1] before TDM reuses it
  }
  #pragma unroll
  for (int cb = 0; cb < 8; ++cb) {
    const int n = cb * 16 + l16;
    #pragma unroll
    for (int r = 0; r < 8; ++r) {
      atomicAdd(&out[((size_t)b * N1 + rowBase + r + 8 * half) * OO + n],
                c[cb][r] * 0.25f);
    }
  }
}

// ---------------------------------------------------------------------------
extern "C" void kernel_launch(void* const* d_in, const int* in_sizes, int n_in,
                              void* d_out, int out_size, void* d_ws, size_t ws_size,
                              hipStream_t stream) {
  const float* x     = (const float*)d_in[0];
  const float* prior = (const float*)d_in[1];
  const unsigned char* xmask = (const unsigned char*)d_in[2];
  const float* Wl    = (const float*)d_in[3];
  const float* wh    = (const float*)d_in[4];
  const float* asrc  = (const float*)d_in[5];
  const float* adst  = (const float*)d_in[6];
  const float* bias  = (const float*)d_in[7];
  float* out = (float*)d_out;

  char* ws = (char*)d_ws;
  __bf16* hp2t = (__bf16*)(ws);                              // 8 MiB (blocked)
  __bf16* hp   = (__bf16*)(ws + (8u << 20));                 // 2 MiB
  float*  ssrc = (float*)(ws + (10u << 20));                 // 128 KiB
  float*  sdst = (float*)(ws + (10u << 20) + 1 * 131072);    // 128 KiB
  float*  sdxb = (float*)(ws + (10u << 20) + 2 * 131072);    // 128 KiB
  float*  mrow = (float*)(ws + (10u << 20) + 3 * 131072);    // 128 KiB
  float*  rz   = (float*)(ws + (10u << 20) + 4 * 131072);    // 128 KiB
  __bf16* whT  = (__bf16*)(ws + (10u << 20) + 5 * 131072);   // 128 KiB

  const int nrows = BB * HH * N1; // 32768

  k_zero   <<<512,  128, 0, stream>>>(ssrc, 2 * nrows);
  k_initout<<<8192, 128, 0, stream>>>(out, bias, BB * N1 * OO);
  k_wht    <<<512,  128, 0, stream>>>(wh, whT);
  k_lin    <<<256,  128, 0, stream>>>(x, Wl, hp);
  k_prior  <<<4,    128, 0, stream>>>(prior, hp);
  k_head   <<<4096, 128, 0, stream>>>(hp, whT, asrc, adst, hp2t, ssrc, sdst);
  k_sdx    <<<nrows / 128, 128, 0, stream>>>(sdst, xmask, sdxb);
  k_stats  <<<nrows / 4,   128, 0, stream>>>(ssrc, sdxb, xmask, mrow, rz);
  k_attn   <<<512,  128, 0, stream>>>(hp2t, ssrc, sdxb, xmask, mrow, rz, out);
}
